// KNIFE_15539191677092
// MI455X (gfx1250) — compile-verified, matrix operand order checked by reading, not run on hardware
//
#include <hip/hip_runtime.h>
#include <hip/hip_bf16.h>

typedef __attribute__((ext_vector_type(16))) _Float16 v16h;
typedef __attribute__((ext_vector_type(8)))  _Float16 v8h;
typedef __attribute__((ext_vector_type(4)))  _Float16 v4h;
typedef __attribute__((ext_vector_type(8)))  float    v8f;

#define NN 8192
#define KK 128
#define DD 128
#define TILES_PER_BLOCK 8
#define SF 132   // fp32 y LDS row stride (floats): 528B -> 4-bank shift/row
#define SHL 136  // f16 L LDS row stride (halfs): 272B -> 4-bank shift/row, 16B aligned
#define SMEM_BYTES (128*SF*4 + 128*SHL*2 + 256*4)
#define LOGC (-117.62413225019811f) /* -64*ln(2*pi) */

// ---------------- kernel 1: per-component scalars ----------------
__global__ __launch_bounds__(128) void gmm_prep(const float* __restrict__ logvar,
                                                const float* __restrict__ weigh,
                                                float* __restrict__ var_ws,
                                                float* __restrict__ bias_ws) {
  __shared__ float red[128];
  const int k = blockIdx.x, t = threadIdx.x;
  float wv = weigh[t];
  red[t] = wv; __syncthreads();
  for (int s = 64; s; s >>= 1) { if (t < s) red[t] = fmaxf(red[t], red[t + s]); __syncthreads(); }
  float m = red[0]; __syncthreads();
  red[t] = __expf(wv - m); __syncthreads();
  for (int s = 64; s; s >>= 1) { if (t < s) red[t] += red[t + s]; __syncthreads(); }
  float lse_w = m + __logf(red[0]); __syncthreads();

  float lv  = logvar[k * DD + t];
  float var = __expf(tanhf(lv));
  var_ws[k * DD + t] = var;
  red[t] = __logf(fabsf(var) + 1e-8f); __syncthreads();
  for (int s = 64; s; s >>= 1) { if (t < s) red[t] += red[t + s]; __syncthreads(); }
  if (t == 0) bias_ws[k] = red[0] + (weigh[k] - lse_w);
}

// ---------------- fragment loaders (wave32, 16x32 f16 A-layout) ----------------
// lanes 0-15: row=lane, K chunks at {0..7} and {16..23}; lanes 16-31: row=lane-16,
// K chunks at {8..15} and {24..31}.  B (32x16) mirrors A with N<->M, so a
// row-major strictly-lower L (rows = output dim i) loads with the same pattern.
__device__ inline v16h ldfragA_f32(const float* y, int row0, int k0) {
  int lane = threadIdx.x & 31;
  const float* p0 = y + (row0 + (lane & 15)) * SF + k0 + (lane >> 4) * 8;
  v16h f;
#pragma unroll
  for (int e = 0; e < 8; ++e) { f[e] = (_Float16)p0[e]; f[e + 8] = (_Float16)p0[16 + e]; }
  return f;
}
__device__ inline v16h ldfragB_f16(const _Float16* Lm, int row0, int k0) {
  int lane = threadIdx.x & 31;
  const _Float16* p0 = Lm + (row0 + (lane & 15)) * SHL + k0 + (lane >> 4) * 8;
  v8h c0 = *(const v8h*)p0;
  v8h c1 = *(const v8h*)(p0 + 16);
  v16h f;
#pragma unroll
  for (int e = 0; e < 8; ++e) { f[e] = c0[e]; f[e + 8] = c1[e]; }
  return f;
}

// ---------------- kernel 2: main batched GEMM + quadratic form ----------------
// Block = (component k, group of 8 n-tiles).  L is staged/masked ONCE per block,
// then the block loops over its n-tiles: stage y -> WMMA -> quadratic form.
__global__ __launch_bounds__(256) void gmm_main(const float* __restrict__ X,
                                                const float* __restrict__ means,
                                                const float* __restrict__ tri,
                                                const float* __restrict__ var_ws,
                                                const float* __restrict__ bias_ws,
                                                float* __restrict__ logits) {
  extern __shared__ char smem_raw[];
  float*    y32 = (float*)smem_raw;                                  // 128 x SF fp32
  _Float16* Lm  = (_Float16*)(smem_raw + 128 * SF * 4);              // 128 x SHL f16
  float*    mv  = (float*)(smem_raw + 128 * SF * 4 + 128 * SHL * 2); // mean[128]
  float*    vv  = mv + 128;                                          // var[128]

  const int tid = threadIdx.x;
  const int k   = blockIdx.x;

  if (tid < 128) mv[tid] = means[k * DD + tid];
  else           vv[tid - 128] = var_ws[k * DD + (tid - 128)];

  // stage strictly-lower-masked L in f16 once (small correction -> f16 safe)
  const float4* T4 = (const float4*)(tri + (size_t)k * (DD * DD));
#pragma unroll
  for (int it = 0; it < 16; ++it) {
    int idx = tid + it * 256;
    int i = idx >> 5, j = (idx & 31) << 2;
    float4 tv = T4[idx];
    v4h hv;
    hv[0] = (_Float16)(((j + 0) < i) ? tv.x : 0.0f);
    hv[1] = (_Float16)(((j + 1) < i) ? tv.y : 0.0f);
    hv[2] = (_Float16)(((j + 2) < i) ? tv.z : 0.0f);
    hv[3] = (_Float16)(((j + 3) < i) ? tv.w : 0.0f);
    *(v4h*)&Lm[i * SHL + j] = hv;
  }
  __syncthreads();

  const int lane  = tid & 31;
  const int strip = (tid >> 5) * 16;            // this wave's 16 samples
  const int sbase = strip + (lane >> 4) * 8;    // D-frag rows held by this lane
  const float biask = bias_ws[k];

  for (int tt = 0; tt < TILES_PER_BLOCK; ++tt) {
    const int n0 = (blockIdx.y * TILES_PER_BLOCK + tt) * 128;

    // stage y = (x - mean) * var in fp32 (exact main term)
    const float4* X4 = (const float4*)(X + (size_t)n0 * DD);
#pragma unroll
    for (int it = 0; it < 16; ++it) {
      int idx = tid + it * 256;          // 0..4095 float4s of the 128x128 tile
      int s = idx >> 5, c = (idx & 31) << 2;
      float4 xv = X4[idx];
      float4 yv;
      yv.x = (xv.x - mv[c + 0]) * vv[c + 0];
      yv.y = (xv.y - mv[c + 1]) * vv[c + 1];
      yv.z = (xv.z - mv[c + 2]) * vv[c + 2];
      yv.w = (xv.w - mv[c + 3]) * vv[c + 3];
      *(float4*)&y32[s * SF + c] = yv;
    }
    __syncthreads();

    v16h afrag[4];                              // A = y_f16, reused across i-tiles
#pragma unroll
    for (int jc = 0; jc < 4; ++jc) afrag[jc] = ldfragA_f32(y32, strip, jc * 32);

    float qp[8] = {0.f, 0.f, 0.f, 0.f, 0.f, 0.f, 0.f, 0.f};
#pragma unroll
    for (int itile = 0; itile < 8; ++itile) {
      v8f acc = {};
#pragma unroll
      for (int jc = 0; jc < 4; ++jc) {
        v16h b = ldfragB_f16(Lm, itile * 16, jc * 32);
        acc = __builtin_amdgcn_wmma_f32_16x16x32_f16(false, afrag[jc], false, b,
                                                     (short)0, acc, false, false);
      }
      int iloc = itile * 16 + (lane & 15);
#pragma unroll
      for (int r = 0; r < 8; ++r) {
        float out = y32[(sbase + r) * SF + iloc] + acc[r]; // fp32 y + f16 corr
        qp[r] += out * out;
      }
    }
    // reduce over the 16 lanes (the i dimension) within each half-wave
#pragma unroll
    for (int m = 1; m <= 8; m <<= 1)
#pragma unroll
      for (int r = 0; r < 8; ++r) qp[r] += __shfl_xor(qp[r], m, 32);

    if ((lane & 15) == 0) {
#pragma unroll
      for (int r = 0; r < 8; ++r)
        logits[(size_t)(n0 + sbase + r) * KK + k] = -0.5f * qp[r] + biask;
    }
    __syncthreads();   // y32 reads done before next tile overwrites it
  }
}

// ---------------- kernel 3: logsumexp over K, partial sums ----------------
__global__ __launch_bounds__(256) void gmm_lse(const float* __restrict__ logits,
                                               float* __restrict__ partials) {
  const int lane = threadIdx.x & 31, wid = threadIdx.x >> 5;
  const int gw = blockIdx.x * 8 + wid;          // 512 waves total
  float acc = 0.f;
  for (int n = gw; n < NN; n += 512) {
    const float* row = logits + (size_t)n * KK;
    float v0 = row[lane], v1 = row[lane + 32], v2 = row[lane + 64], v3 = row[lane + 96];
    float m = fmaxf(fmaxf(v0, v1), fmaxf(v2, v3));
    for (int x = 16; x; x >>= 1) m = fmaxf(m, __shfl_xor(m, x, 32));
    float s = __expf(v0 - m) + __expf(v1 - m) + __expf(v2 - m) + __expf(v3 - m);
    for (int x = 16; x; x >>= 1) s += __shfl_xor(s, x, 32);
    acc += -(LOGC + (m + __logf(s)));
  }
  __shared__ float wsum[8];
  if (lane == 0) wsum[wid] = acc;
  __syncthreads();
  if (threadIdx.x == 0) {
    float b = 0.f;
    for (int w = 0; w < 8; ++w) b += wsum[w];
    partials[blockIdx.x] = b;
  }
}

__global__ __launch_bounds__(64) void gmm_finish(const float* __restrict__ partials,
                                                 float* __restrict__ out) {
  __shared__ float red[64];
  const int t = threadIdx.x;
  red[t] = partials[t]; __syncthreads();
  for (int s = 32; s; s >>= 1) { if (t < s) red[t] += red[t + s]; __syncthreads(); }
  if (t == 0) out[0] = red[0] * (1.0f / (float)NN);
}

// ---------------- host entry ----------------
extern "C" void kernel_launch(void* const* d_in, const int* in_sizes, int n_in,
                              void* d_out, int out_size, void* d_ws, size_t ws_size,
                              hipStream_t stream) {
  const float* X      = (const float*)d_in[0];
  const float* means  = (const float*)d_in[1];
  const float* logvar = (const float*)d_in[2];
  const float* tri    = (const float*)d_in[3];
  const float* weigh  = (const float*)d_in[4];

  float* ws       = (float*)d_ws;
  float* logits   = ws;                                  // N*K fp32 (4 MB)
  float* var_ws   = ws + (size_t)NN * KK;                // K*D
  float* bias_ws  = var_ws + (size_t)KK * DD;            // K
  float* partials = bias_ws + KK;                        // 64

  hipFuncSetAttribute(reinterpret_cast<const void*>(gmm_main),
                      hipFuncAttributeMaxDynamicSharedMemorySize, SMEM_BYTES);

  gmm_prep<<<KK, 128, 0, stream>>>(logvar, weigh, var_ws, bias_ws);
  gmm_main<<<dim3(KK, NN / (128 * TILES_PER_BLOCK)), 256, SMEM_BYTES, stream>>>(
      X, means, tri, var_ws, bias_ws, logits);
  gmm_lse<<<64, 256, 0, stream>>>(logits, partials);
  gmm_finish<<<1, 64, 0, stream>>>(partials, (float*)d_out);
}
